// Attention_11579231830088
// MI455X (gfx1250) — matrix operand order repaired and checked by reading, not compiled, not run on hardware
//
#include <hip/hip_runtime.h>
#include <stdint.h>

// Problem constants (fixed by the reference)
#define B_      8
#define LQ      2048
#define LK      2048
#define DD      1024
#define MT      32        // query tile rows per workgroup
#define NT      32        // key tile cols per iteration
#define WAVES   8
#define DSLICE  128       // D columns owned per wave (8*128 = 1024)
#define THREADS 256
#define VT_STRIDE 40      // padded row stride (bf16 elems) for transposed V tile
#define NEG_INF_F (-1e10f)

typedef __attribute__((ext_vector_type(16))) __bf16 v16bf;
typedef __attribute__((ext_vector_type(8)))  float  v8f;

struct U32x8 { uint4 a, b; };

static __device__ __forceinline__ v16bf as_v16bf(uint4 a, uint4 b) {
  union { U32x8 u; v16bf v; } t;
  t.u.a = a; t.u.b = b;
  return t.v;
}

static __device__ __forceinline__ v8f zero8() {
  v8f z = {0.f, 0.f, 0.f, 0.f, 0.f, 0.f, 0.f, 0.f};
  return z;
}

// round-to-nearest-even f32 -> bf16 bits
static __device__ __forceinline__ unsigned short f2bf(float f) {
  unsigned int u = __float_as_uint(f);
  u += 0x7fffu + ((u >> 16) & 1u);
  return (unsigned short)(u >> 16);
}

__global__ __launch_bounds__(THREADS)
void fattn_wmma_kernel(const float* __restrict__ hidden,
                       const float* __restrict__ keys,
                       const float* __restrict__ values,
                       const int*   __restrict__ mask,
                       float* __restrict__ out) {
  const int tid  = threadIdx.x;
  const int wave = tid >> 5;
  const int lane = tid & 31;
  const int h    = lane >> 4;   // lane-half (WMMA layout selector)
  const int ln   = lane & 15;
  const int b    = blockIdx.y;
  const int q0   = blockIdx.x * MT;
  const int dbase = wave * DSLICE;

  // -------- dynamic LDS carving (≈215 KB, legal on CDNA5: 320 KB/WGP) --------
  extern __shared__ char smem[];
  unsigned short* sQ  = (unsigned short*)smem;          // 32 x 1024 bf16 (64 KB)
  unsigned short* sK  = sQ + MT * DD;                   // 32 x 1024 bf16 (64 KB)
  unsigned short* sVt = sK + NT * DD;                   // 1024 x 40 bf16 (80 KB), [d][key] transposed
  float*          sS  = (float*)(sVt + DD * VT_STRIDE); // 32 x 32 f32 score tile
  unsigned short* sP  = (unsigned short*)(sS + MT * NT);// 32 x 32 bf16 probs
  float*          sM  = (float*)(sP + MT * NT);         // running row max
  float*          sL  = sM + MT;                        // running row sum
  float*          sScale = sL + MT;                     // per-row rescale factor
  int*            sMask  = (int*)(sScale + MT);         // 32 key-mask ints
  int*            sFlag  = sMask + NT;                  // "any row max updated" flag

  // -------- load Q tile once: f32 -> bf16 into LDS (coalesced float4) --------
  {
    const float* gq = hidden + ((size_t)b * LQ + q0) * DD;
    for (int i = tid; i < MT * DD / 4; i += THREADS) {
      int row = i >> 8;              // 256 float4 per row
      int c4  = i & 255;
      float4 v = ((const float4*)(gq + (size_t)row * DD))[c4];
      uint2 p;
      p.x = (unsigned)f2bf(v.x) | ((unsigned)f2bf(v.y) << 16);
      p.y = (unsigned)f2bf(v.z) | ((unsigned)f2bf(v.w) << 16);
      *(uint2*)(sQ + (size_t)row * DD + c4 * 4) = p;
    }
  }
  if (tid < MT) { sM[tid] = -1e30f; sL[tid] = 0.f; }

  // O accumulator: this wave owns O[32][dbase .. dbase+127] as 2x8 WMMA tiles
  v8f accO[2][8];
#pragma unroll
  for (int i = 0; i < 2; ++i)
#pragma unroll
    for (int c = 0; c < 8; ++c) accO[i][c] = zero8();

  // ======================= online-softmax key loop =======================
  for (int kt = 0; kt < LK / NT; ++kt) {
    const int k0 = kt * NT;
    __syncthreads();  // previous iteration done with sK/sVt/sP/sS/sFlag

    const float* gk = keys   + ((size_t)b * LK + k0) * DD;
    const float* gv = values + ((size_t)b * LK + k0) * DD;

    // ---- stage K tile (bf16) and V tile (bf16, transposed) ----
    for (int i = tid; i < NT * DD / 4; i += THREADS) {
      int row = i >> 8;
      int c4  = i & 255;
      float4 v = ((const float4*)(gk + (size_t)row * DD))[c4];
      uint2 p;
      p.x = (unsigned)f2bf(v.x) | ((unsigned)f2bf(v.y) << 16);
      p.y = (unsigned)f2bf(v.z) | ((unsigned)f2bf(v.w) << 16);
      *(uint2*)(sK + (size_t)row * DD + c4 * 4) = p;

      float4 w = ((const float4*)(gv + (size_t)row * DD))[c4];
      int d = c4 * 4;
      sVt[(d + 0) * VT_STRIDE + row] = f2bf(w.x);
      sVt[(d + 1) * VT_STRIDE + row] = f2bf(w.y);
      sVt[(d + 2) * VT_STRIDE + row] = f2bf(w.z);
      sVt[(d + 3) * VT_STRIDE + row] = f2bf(w.w);
    }
    for (int i = tid; i < MT * NT; i += THREADS) sS[i] = 0.f;
    if (tid == 0) *sFlag = 0;

    // ---- async copy of the key-mask tile into LDS (CDNA5 async path) ----
    if (tid < 32) {
      unsigned ldsdst = (unsigned)(size_t)sMask + (unsigned)lane * 4u;
      unsigned voff   = (unsigned)lane * 4u;
      const int* gsrc = mask + (size_t)b * LK + k0;
      asm volatile(
          "global_load_async_to_lds_b32 %0, %1, %2\n\t"
          "s_wait_asynccnt 0"
          :: "v"(ldsdst), "v"(voff), "s"(gsrc)
          : "memory");
    }

    // prefetch next tiles into cache hierarchy (global_prefetch_b8)
    if (kt + 1 < LK / NT) {
      __builtin_prefetch(gk + (size_t)NT * DD + (size_t)tid * 128, 0, 0);
      __builtin_prefetch(gv + (size_t)NT * DD + (size_t)tid * 128, 0, 0);
    }
    __syncthreads();

    // ---- S_partial = Q_slice * K_sliceᵀ  (this wave's 128-wide D slice) ----
    v8f accS[2][2];
#pragma unroll
    for (int i = 0; i < 2; ++i)
#pragma unroll
      for (int j = 0; j < 2; ++j) accS[i][j] = zero8();

#pragma unroll
    for (int kk = 0; kk < 4; ++kk) {
      const int dc = dbase + kk * 32;
      // A (Q rows): per-lane elems j -> K=16h+j (contiguous)
      v16bf aQ[2];
#pragma unroll
      for (int i = 0; i < 2; ++i) {
        const unsigned short* base = sQ + (size_t)(16 * i + ln) * DD + dc + 16 * h;
        uint4 lo = *(const uint4*)base;
        uint4 hi = *(const uint4*)(base + 8);   // +8 shorts = K+8
        aQ[i] = as_v16bf(lo, hi);
      }
      // B (Kᵀ): per-lane elems j -> Kdim = 16h + j (contiguous 32B), N = key = ln
      v16bf bK[2];
#pragma unroll
      for (int j = 0; j < 2; ++j) {
        const unsigned short* base = sK + (size_t)(16 * j + ln) * DD + dc + 16 * h;
        uint4 lo = *(const uint4*)base;
        uint4 hi = *(const uint4*)(base + 8);
        bK[j] = as_v16bf(lo, hi);
      }
#pragma unroll
      for (int i = 0; i < 2; ++i)
#pragma unroll
        for (int j = 0; j < 2; ++j)
          accS[i][j] = __builtin_amdgcn_wmma_f32_16x16x32_bf16(
              false, aQ[i], false, bK[j], (short)0, accS[i][j], false, false);
    }

    // ---- cross-wave reduction of score partials via LDS float atomics ----
#pragma unroll
    for (int i = 0; i < 2; ++i)
#pragma unroll
      for (int j = 0; j < 2; ++j)
#pragma unroll
        for (int r = 0; r < 8; ++r) {
          int m = 16 * i + 8 * h + r;
          int n = 16 * j + ln;
          atomicAdd(&sS[m * NT + n], accS[i][j][r]);
        }
    __syncthreads();

    // ---- masked online softmax update, parallel over all 8 waves ----
    // wave w owns rows 4w..4w+3; 8 lanes per row, 4 columns per lane.
    {
      const int row = 4 * wave + (lane >> 3);
      const int cj  = lane & 7;
      const float mprev = sM[row];
      float sc[4];
      float lm = -3.0e38f;
#pragma unroll
      for (int t = 0; t < 4; ++t) {
        const int n = cj + 8 * t;
        float s = sS[row * NT + n];
        if (sMask[n] == 0) s = NEG_INF_F;
        sc[t] = s;
        lm = fmaxf(lm, s);
      }
#pragma unroll
      for (int off = 1; off < 8; off <<= 1)
        lm = fmaxf(lm, __shfl_xor(lm, off));
      const float mnew = fmaxf(mprev, lm);
      const float corr = __expf(mprev - mnew);
      float psum = 0.f;
#pragma unroll
      for (int t = 0; t < 4; ++t) {
        const int n = cj + 8 * t;
        float p = __expf(sc[t] - mnew);
        psum += p;
        sP[row * NT + n] = f2bf(p);
      }
#pragma unroll
      for (int off = 1; off < 8; off <<= 1)
        psum += __shfl_xor(psum, off);
      if (cj == 0) {
        sL[row] = sL[row] * corr + psum;
        sM[row] = mnew;
        sScale[row] = corr;
        if (mnew > mprev) *sFlag = 1;   // benign multi-writer of identical value
      }
    }
    __syncthreads();

    // ---- rescale O accumulator only when some row max advanced ----
    if (*sFlag) {
      float scl[2][8];
#pragma unroll
      for (int i = 0; i < 2; ++i)
#pragma unroll
        for (int r = 0; r < 8; ++r) scl[i][r] = sScale[16 * i + 8 * h + r];
#pragma unroll
      for (int i = 0; i < 2; ++i)
#pragma unroll
        for (int c = 0; c < 8; ++c)
#pragma unroll
          for (int r = 0; r < 8; ++r) accO[i][c][r] *= scl[i][r];
    }

    // ---- O += P * V_slice ----
    v16bf aP[2];
#pragma unroll
    for (int i = 0; i < 2; ++i) {
      const unsigned short* base = sP + (size_t)(16 * i + ln) * NT + 16 * h;
      uint4 lo = *(const uint4*)base;
      uint4 hi = *(const uint4*)(base + 8);
      aP[i] = as_v16bf(lo, hi);
    }
#pragma unroll
    for (int c = 0; c < 8; ++c) {
      const int dcol = dbase + c * 16 + ln;   // this lane's V column (N index)
      const unsigned short* base = sVt + (size_t)dcol * VT_STRIDE + 16 * h;
      uint4 lo = *(const uint4*)base;
      uint4 hi = *(const uint4*)(base + 8);
      v16bf bV = as_v16bf(lo, hi);
#pragma unroll
      for (int i = 0; i < 2; ++i)
        accO[i][c] = __builtin_amdgcn_wmma_f32_16x16x32_bf16(
            false, aP[i], false, bV, (short)0, accO[i][c], false, false);
    }
  }

  // ======================= epilogue: O / l ->  global =======================
  __syncthreads();
  float linv[2][8];
#pragma unroll
  for (int i = 0; i < 2; ++i)
#pragma unroll
    for (int r = 0; r < 8; ++r) linv[i][r] = 1.f / sL[16 * i + 8 * h + r];

  float* go = out + ((size_t)b * LQ + q0) * DD + dbase;
#pragma unroll
  for (int i = 0; i < 2; ++i)
#pragma unroll
    for (int r = 0; r < 8; ++r) {
      const int m = 16 * i + 8 * h + r;
      float* rowp = go + (size_t)m * DD;
#pragma unroll
      for (int c = 0; c < 8; ++c)
        rowp[c * 16 + ln] = accO[i][c][r] * linv[i][r];
    }
}

extern "C" void kernel_launch(void* const* d_in, const int* in_sizes, int n_in,
                              void* d_out, int out_size, void* d_ws, size_t ws_size,
                              hipStream_t stream) {
  (void)in_sizes; (void)n_in; (void)out_size; (void)d_ws; (void)ws_size;
  const float* hidden = (const float*)d_in[0];
  const float* keys   = (const float*)d_in[1];
  const float* values = (const float*)d_in[2];
  const int*   mask   = (const int*)d_in[3];
  float*       out    = (float*)d_out;

  // dynamic LDS: sQ + sK + sVt + sS + sP + stats + mask + flag
  size_t smem = (size_t)(MT * DD) * 2        // sQ
              + (size_t)(NT * DD) * 2        // sK
              + (size_t)(DD * VT_STRIDE) * 2 // sVt
              + (size_t)(MT * NT) * 4        // sS
              + (size_t)(MT * NT) * 2        // sP
              + (size_t)(3 * MT) * 4         // sM, sL, sScale
              + (size_t)NT * 4               // sMask
              + 16;                          // sFlag (+pad)

  dim3 grid(LQ / MT, B_);
  dim3 block(THREADS);
  fattn_wmma_kernel<<<grid, block, smem, stream>>>(hidden, keys, values, mask, out);
}
